// EnformerDotProductSelfAttention_27178553049153
// MI455X (gfx1250) — matrix operand order repaired and checked by reading, not compiled, not run on hardware
//
#include <hip/hip_runtime.h>
#include <hip/hip_bf16.h>

// Problem constants (B=1)
#define SEQ   2048
#define NH    8
#define HD    64
#define NPE   64
#define HALFN 32
#define KSTR  34   // 33 suffix-sum entries, padded

typedef __attribute__((ext_vector_type(16))) __bf16 v16bf;
typedef __attribute__((ext_vector_type(8)))  float  v8f;

union AFrag { v16bf v; uint4 u[2]; };

__device__ __forceinline__ v8f wmma_bf16(v16bf a, v16bf b, v8f c) {
  // D = A(16x32 bf16) * B(32x16 bf16) + C(16x16 f32)
  return __builtin_amdgcn_wmma_f32_16x16x32_bf16(false, a, false, b, (short)0, c, false, false);
}

// Load a 16x32 bf16 A-fragment strip (lane half selects K 0-7/8-15 within each
// 16-wide K group): u[0] <- K = half*8 .. +7, u[1] <- K = 16 + half*8 .. +7
__device__ __forceinline__ void load_afrag(AFrag& f, const __bf16* rowp, int half) {
  f.u[0] = *(const uint4*)(rowp + half * 8);
  f.u[1] = *(const uint4*)(rowp + 16 + half * 8);
}

// ---------------------------------------------------------------------------
// Prep 1: bf16 Q [H][S][D], bf16 K [H][S][D], bf16 V^T [H][D][S], uk[h][j]
// ---------------------------------------------------------------------------
__global__ __launch_bounds__(256) void prep_qkv(
    const float* __restrict__ q, const float* __restrict__ k, const float* __restrict__ v,
    const float* __restrict__ u,
    __bf16* __restrict__ Qb, __bf16* __restrict__ Kb, __bf16* __restrict__ Vt,
    float* __restrict__ uk)
{
  int r = blockIdx.x * 256 + threadIdx.x;   // 0 .. NH*SEQ-1
  int h = r / SEQ, s = r % SEQ;
  const float* qr = q + ((size_t)s * NH + h) * HD;
  const float* kr = k + ((size_t)s * NH + h) * HD;
  const float* vr = v + ((size_t)s * NH + h) * HD;
  const float* ur = u + h * HD;
  __bf16* qo = Qb + ((size_t)h * SEQ + s) * HD;
  __bf16* ko = Kb + ((size_t)h * SEQ + s) * HD;
  float acc = 0.f;
  #pragma unroll 8
  for (int d = 0; d < HD; ++d) {
    float kv = kr[d];
    qo[d] = (__bf16)qr[d];
    ko[d] = (__bf16)kv;
    acc += ur[d] * kv;
    Vt[((size_t)h * HD + d) * SEQ + s] = (__bf16)vr[d];
  }
  uk[h * SEQ + s] = acc;
}

// ---------------------------------------------------------------------------
// Prep 2: wq[i,h,:] = q[i,h,:] @ w[h] ; suffix sums over n:
//   suss[h][i][k] = ( sum_{n>=k} wq[n],  sum_{n>=k} wq[32+n] ),  k = 0..32
// ---------------------------------------------------------------------------
__global__ __launch_bounds__(128) void prep_wq(
    const float* __restrict__ q, const float* __restrict__ w,
    float2* __restrict__ suss)
{
  int r = blockIdx.x * 128 + threadIdx.x;
  int h = r / SEQ, i = r % SEQ;
  const float* qr = q + ((size_t)i * NH + h) * HD;
  const float* wh = w + (size_t)h * HD * NPE;
  float acc[NPE];
  #pragma unroll
  for (int n = 0; n < NPE; ++n) acc[n] = 0.f;
  for (int d = 0; d < HD; ++d) {
    float qd = qr[d];
    const float* wr = wh + (size_t)d * NPE;
    #pragma unroll
    for (int n = 0; n < NPE; ++n) acc[n] += qd * wr[n];
  }
  float2* o = suss + ((size_t)h * SEQ + i) * KSTR;
  float su = 0.f, ss = 0.f;
  o[HALFN] = make_float2(0.f, 0.f);
  for (int kk = HALFN - 1; kk >= 0; --kk) {
    su += acc[kk];
    ss += acc[HALFN + kk];
    o[kk] = make_float2(su, ss);
  }
}

// ---------------------------------------------------------------------------
// Prep 3: center widths cw_n = pow_rate^(n+1); LUT k(|d|) = #{n: cw_n < |d|};
//         vw[h][n] = v[h]·w[h,:,n]; vus[h][k] = suffix sums of vw
// ---------------------------------------------------------------------------
__global__ __launch_bounds__(64) void prep_small(
    const float* __restrict__ v, const float* __restrict__ w,
    float2* __restrict__ vus, unsigned char* __restrict__ lut)
{
  __shared__ float cw[HALFN];
  __shared__ float vw[NH][NPE];
  int t = threadIdx.x;
  if (t < HALFN) {
    float pr = expf(logf((float)(SEQ + 1) * 0.5f) / (float)HALFN);
    cw[t] = powf(pr, (float)(t + 1));
  }
  __syncthreads();
  for (int ad = t; ad < SEQ; ad += 64) {
    int kk = 0;
    #pragma unroll
    for (int n = 0; n < HALFN; ++n) kk += (cw[n] < (float)ad) ? 1 : 0;
    lut[ad] = (unsigned char)kk;
  }
  for (int idx = t; idx < NH * NPE; idx += 64) {
    int h = idx / NPE, n = idx % NPE;
    float s = 0.f;
    for (int d = 0; d < HD; ++d) s += v[h * HD + d] * w[((size_t)h * HD + d) * NPE + n];
    vw[h][n] = s;
  }
  __syncthreads();
  if (t < NH) {
    float su = 0.f, ss = 0.f;
    vus[t * KSTR + HALFN] = make_float2(0.f, 0.f);
    for (int kk = HALFN - 1; kk >= 0; --kk) {
      su += vw[t][kk];
      ss += vw[t][HALFN + kk];
      vus[t * KSTR + kk] = make_float2(su, ss);
    }
  }
}

// ---------------------------------------------------------------------------
// Flash attention with bf16 WMMA, logits computed transposed:
//   S^T(16j x 16i) = K_tile(A) x Q^T(B);  O^T(16d x 16i) += V^T(A) x P^T(B)
// C/D layout: N = lane%16 = query row i  ->  softmax stats are per-lane scalars.
// One wave = 16 query rows; block = 4 waves = 64 rows; grid = H * S/64.
// K fragments are register double-buffered (loaded one iteration ahead); V
// fragments are issued right after the S^T WMMAs so the ~100-instruction
// bias/softmax VALU block hides their latency.
// ---------------------------------------------------------------------------
__global__ __launch_bounds__(128) void attn_fwd(
    const __bf16* __restrict__ Qb, const __bf16* __restrict__ Kb, const __bf16* __restrict__ Vt,
    const float2* __restrict__ suss_g, const float* __restrict__ uk_g,
    const float2* __restrict__ vus_g, const unsigned char* __restrict__ lut_g,
    float* __restrict__ out)
{
  __shared__ unsigned char lut[SEQ];
  __shared__ float uks[SEQ];
  __shared__ float2 suss[4][16][KSTR];

  const int h    = blockIdx.x >> 5;
  const int it   = blockIdx.x & 31;
  const int tid  = threadIdx.x;
  const int wv   = tid >> 5;
  const int lane = tid & 31;
  const int row  = lane & 15;
  const int half = lane >> 4;
  const int i0   = it * 64 + wv * 16;
  const int i    = i0 + row;

  for (int x = tid; x < SEQ; x += 128) {
    lut[x] = lut_g[x];
    uks[x] = uk_g[h * SEQ + x];
  }
  // fold per-head vr suffix sums into per-row q suffix sums
  const float2* vus = vus_g + h * KSTR;
  for (int x = lane; x < 16 * KSTR; x += 32) {   // 544/32 = 17 exact
    int rr = x / KSTR, kk = x % KSTR;
    float2 a = suss_g[((size_t)h * SEQ + i0 + rr) * KSTR + kk];
    float2 b = vus[kk];
    suss[wv][rr][kk] = make_float2(a.x + b.x, a.y + b.y);
  }
  __syncthreads();

  // Q^T B-fragments: u[0] <- d = half*8 .. +7, u[1] <- d = 16 + half*8 .. +7 within each K=32 slab
  const __bf16* qrow = Qb + ((size_t)h * SEQ + i) * HD;
  AFrag qlo, qhi;
  qlo.u[0] = *(const uint4*)(qrow + half * 8);
  qlo.u[1] = *(const uint4*)(qrow + 16 + half * 8);
  qhi.u[0] = *(const uint4*)(qrow + 32 + half * 8);
  qhi.u[1] = *(const uint4*)(qrow + 32 + 16 + half * 8);

  const __bf16* kbase = Kb + (size_t)h * SEQ * HD;
  const __bf16* vbase = Vt + (size_t)h * HD * SEQ;

  v8f o0 = {}, o1 = {}, o2 = {}, o3 = {};
  float m = -3.0e38f, lsum = 0.f;
  const float L2E = 1.4426950408889634f;

  // --- pipeline prologue: K A-fragments for j0 = 0 ---
  AFrag c0lo, c0hi, c1lo, c1hi;
  {
    const __bf16* kr0 = kbase + (size_t)row * HD;
    const __bf16* kr1 = kbase + (size_t)(16 + row) * HD;
    load_afrag(c0lo, kr0, half);
    load_afrag(c0hi, kr0 + 32, half);
    load_afrag(c1lo, kr1, half);
    load_afrag(c1hi, kr1 + 32, half);
  }

  for (int j0 = 0; j0 < SEQ; j0 += 32) {
    // S^T for this iteration (K frags already resident)
    v8f z = {};
    v8f st0 = wmma_bf16(c0lo.v, qlo.v, wmma_bf16(c0hi.v, qhi.v, z));
    v8f st1 = wmma_bf16(c1lo.v, qlo.v, wmma_bf16(c1hi.v, qhi.v, z));

    // issue V^T fragments for this iteration early (consumed after softmax)
    AFrag vf0, vf1, vf2, vf3;
    load_afrag(vf0, vbase + (size_t)(row)      * SEQ + j0, half);
    load_afrag(vf1, vbase + (size_t)(16 + row) * SEQ + j0, half);
    load_afrag(vf2, vbase + (size_t)(32 + row) * SEQ + j0, half);
    load_afrag(vf3, vbase + (size_t)(48 + row) * SEQ + j0, half);

    // issue next iteration's K fragments (wrap to 0 on last iter: harmless)
    AFrag n0lo, n0hi, n1lo, n1hi;
    {
      int jn = (j0 + 32 < SEQ) ? (j0 + 32) : 0;
      const __bf16* kr0 = kbase + (size_t)(jn + row) * HD;
      const __bf16* kr1 = kbase + (size_t)(jn + 16 + row) * HD;
      load_afrag(n0lo, kr0, half);
      load_afrag(n0hi, kr0 + 32, half);
      load_afrag(n1lo, kr1, half);
      load_afrag(n1hi, kr1 + 32, half);
      __builtin_prefetch(kr0 + 32 * HD, 0, 1);   // j0+64 -> global_prefetch
      __builtin_prefetch(kr1 + 32 * HD, 0, 1);
    }

    // bias: logit = (qk + SU'[i][k] + sgn*SS'[i][k] + uk[j]) / sqrt(D)
    float v0[8], v1[8];
    float tmax = -3.0e38f;
    #pragma unroll
    for (int p = 0; p < 8; ++p) {
      int jl = p + 8 * half;               // S^T C-layout: M(row)=j-local
      {
        int j = j0 + jl;
        int dd = i - j; int ad = dd < 0 ? -dd : dd;
        float sg = (float)((dd > 0) - (dd < 0));
        float2 e = suss[wv][row][lut[ad]];
        v0[p] = (st0[p] + e.x + sg * e.y + uks[j]) * 0.125f;
      }
      {
        int j = j0 + 16 + jl;
        int dd = i - j; int ad = dd < 0 ? -dd : dd;
        float sg = (float)((dd > 0) - (dd < 0));
        float2 e = suss[wv][row][lut[ad]];
        v1[p] = (st1[p] + e.x + sg * e.y + uks[j]) * 0.125f;
      }
      tmax = fmaxf(tmax, fmaxf(v0[p], v1[p]));
    }
    // combine lane halves (same query row i lives in lane and lane^16)
    tmax = fmaxf(tmax, __shfl_xor(tmax, 16, 32));
    float nm = fmaxf(m, tmax);
    float alpha = exp2f((m - nm) * L2E);
    float ts = 0.f;
    float p0[8], p1[8];
    #pragma unroll
    for (int p = 0; p < 8; ++p) {
      p0[p] = exp2f((v0[p] - nm) * L2E);
      p1[p] = exp2f((v1[p] - nm) * L2E);
      ts += p0[p] + p1[p];
    }
    ts += __shfl_xor(ts, 16, 32);
    lsum = lsum * alpha + ts;
    m = nm;
    #pragma unroll
    for (int p = 0; p < 8; ++p) { o0[p] *= alpha; o1[p] *= alpha; o2[p] *= alpha; o3[p] *= alpha; }

    // Build P^T B-fragment: element p holds j-local = half*8 + p, element p+8 holds 16 + half*8 + p.
    // Own lane has j = p + 8*half (tile0) and 16 + p + 8*half (tile1).
    v16bf pb;
    #pragma unroll
    for (int p = 0; p < 8; ++p) {
      float x0 = __shfl_xor(p0[p], 16, 32);
      float x1 = __shfl_xor(p1[p], 16, 32);
      float lo = p0[p];   // j-local = half*8 + p
      float hi = p1[p];   // j-local = 16 + half*8 + p
      pb[p]     = (__bf16)lo;
      pb[p + 8] = (__bf16)hi;
    }

    // O^T += V^T(A: 16 d-rows x 32 j) x P^T(B)
    o0 = wmma_bf16(vf0.v, pb, o0);
    o1 = wmma_bf16(vf1.v, pb, o1);
    o2 = wmma_bf16(vf2.v, pb, o2);
    o3 = wmma_bf16(vf3.v, pb, o3);

    // rotate K double buffer
    c0lo = n0lo; c0hi = n0hi; c1lo = n1lo; c1hi = n1hi;
  }

  // epilogue: O^T C-layout -> out (B,S,H,D) f32, divide by row sum
  float rl = 1.0f / lsum;
  float* obase = out + ((size_t)i * NH + h) * HD;
  #pragma unroll
  for (int p = 0; p < 8; ++p) {
    int dl = p + 8 * half;
    obase[dl]      = o0[p] * rl;
    obase[16 + dl] = o1[p] * rl;
    obase[32 + dl] = o2[p] * rl;
    obase[48 + dl] = o3[p] * rl;
  }
}

// ---------------------------------------------------------------------------
extern "C" void kernel_launch(void* const* d_in, const int* in_sizes, int n_in,
                              void* d_out, int out_size, void* d_ws, size_t ws_size,
                              hipStream_t stream) {
  const float* q  = (const float*)d_in[0];
  const float* k  = (const float*)d_in[1];
  const float* v  = (const float*)d_in[2];
  const float* u  = (const float*)d_in[3];
  const float* vv = (const float*)d_in[4];
  const float* w  = (const float*)d_in[5];
  float* out = (float*)d_out;

  char* ws = (char*)d_ws;
  size_t off = 0;
  auto carve = [&](size_t bytes) -> void* {
    void* p = ws + off;
    off += (bytes + 255) & ~(size_t)255;
    return p;
  };
  __bf16* Qb = (__bf16*)carve((size_t)NH * SEQ * HD * 2);   // 2 MB
  __bf16* Kb = (__bf16*)carve((size_t)NH * SEQ * HD * 2);   // 2 MB
  __bf16* Vt = (__bf16*)carve((size_t)NH * HD * SEQ * 2);   // 2 MB
  float2* suss = (float2*)carve((size_t)NH * SEQ * KSTR * sizeof(float2)); // 4.5 MB
  float*  uk   = (float*)carve((size_t)NH * SEQ * sizeof(float));
  float2* vus  = (float2*)carve((size_t)NH * KSTR * sizeof(float2));
  unsigned char* lut = (unsigned char*)carve((size_t)SEQ);

  prep_qkv<<<NH * SEQ / 256, 256, 0, stream>>>(q, k, v, u, Qb, Kb, Vt, uk);
  prep_wq<<<NH * SEQ / 128, 128, 0, stream>>>(q, w, suss);
  prep_small<<<1, 64, 0, stream>>>(vv, w, vus, lut);
  attn_fwd<<<NH * (SEQ / 64), 128, 0, stream>>>(Qb, Kb, Vt, suss, uk, vus, lut, out);
}
